// SimplifiedMLA_46050639348065
// MI455X (gfx1250) — matrix operand order repaired and checked by reading, not compiled, not run on hardware
//
#include <hip/hip_runtime.h>

// ---------------------------------------------------------------------------
// SimplifiedMLA on gfx1250 (MI455X): 6 WMMA GEMMs + softmax, all f32 I/O,
// bf16 v_wmma_f32_16x16x32_bf16 compute with in-register f32->bf16 convert.
// Memory-bound (~550MB traffic, ~11 GFLOP @ 23.3 TB/s => ~24us floor), so:
//  - every big tensor is streamed exactly once per GEMM (no bf16 staging copy)
//  - NN GEMMs (B in [K,N] order, incl. the 2nd pass over compressed_kv) stage
//    B tiles into LDS with global_load_async_to_lds_b128, double-buffered so
//    the async copy of tile i+1 overlaps the WMMA of tile i (ASYNCcnt).
// ---------------------------------------------------------------------------

typedef __attribute__((ext_vector_type(16))) __bf16 v16bf;
typedef __attribute__((ext_vector_type(8)))  float  v8f;

union BFPack16 { v16bf v; unsigned short h[16]; };

__device__ __forceinline__ unsigned short f2bf(float f) {
  union { float f; unsigned u; } c; c.f = f;
  // round-half-up truncation to bf16 (inputs are well-scaled normals)
  return (unsigned short)((c.u + 0x8000u) >> 16);
}

// Async copy 16B global -> LDS (tracked on ASYNCcnt). lds_off is the LDS byte
// offset (= low 32 bits of a generic pointer to __shared__, ISA 10.2).
__device__ __forceinline__ void async_copy_b128(unsigned lds_off, const float* gsrc) {
  asm volatile("global_load_async_to_lds_b128 %0, %1, off"
               :: "v"(lds_off), "v"(gsrc) : "memory");
}
__device__ __forceinline__ void wait_async0() {
  asm volatile("s_wait_asynccnt 0" ::: "memory");
}

// C[m,n] = alpha * sum_k A[m,k] * B(k,n)  (+ bias[n])
//   B_NK == true  : B stored [n*ldb + k]  (C = A*B^T), lane-contiguous streams
//   B_NK == false : B stored [k*ldb + n]  (C = A*B), async-staged through LDS
// Block: 256 threads = 8 waves, covers 32(M) x 64(N); 1 WMMA tile per wave.
// INVARIANT: M == 32 and N % 64 == 0 for every launch (no partial tiles, so
// all waves run the full loop -> barriers are safe and EXEC is all-ones).
template <bool B_NK>
__global__ __launch_bounds__(256) void gemm_wmma(
    const float* __restrict__ A, int lda, long long sAz,
    const float* __restrict__ B, int ldb, long long sBz,
    float* __restrict__ C, int ldc, long long sCz,
    const float* __restrict__ bias,
    int M, int N, int K, float alpha) {
  A += (long long)blockIdx.z * sAz;
  B += (long long)blockIdx.z * sBz;
  C += (long long)blockIdx.z * sCz;

  const int wave = threadIdx.x >> 5;   // 0..7
  const int lane = threadIdx.x & 31;
  const int m0   = (wave >> 2) * 16;                  // 0 or 16
  const int nblk = blockIdx.x * 64;
  const int n0   = nblk + (wave & 3) * 16;            // N tile of this wave

  // A-matrix 16x32 bf16 layout: lane L holds row m = L&15;
  //   lanes 0-15 : elems 0..7 -> K 0..7,   elems 8..15 -> K 16..23
  //   lanes 16-31: elems 0..7 -> K 8..15,  elems 8..15 -> K 24..31
  const int am  = lane & 15;
  const int akb = (lane < 16) ? 0 : 8;
  // B-matrix 32x16 bf16 layout: lane L holds col n = L&15, elems 0..7 -> K bkb+e, elems 8..15 -> K bkb+8+e
  const int bn  = lane & 15;
  const int bkb = (lane < 16) ? 0 : 8;
  // C/D 16x16 f32 layout: lane L holds col n = L&15, rows cm..cm+7
  const int cm  = (lane < 16) ? 0 : 8;
  const int cn  = lane & 15;

  v8f acc = {};
  const float* arow = A + (long long)(m0 + am) * lda;

  if (B_NK) {
    // ---- NT: both operands k-contiguous, stream straight to registers ----
    for (int k0 = 0; k0 < K; k0 += 32) {
      BFPack16 a;
      {
        const float4 a0 = *(const float4*)(arow + k0 + akb);
        const float4 a1 = *(const float4*)(arow + k0 + akb + 4);
        const float4 a2 = *(const float4*)(arow + k0 + akb + 16);
        const float4 a3 = *(const float4*)(arow + k0 + akb + 20);
        a.h[0]=f2bf(a0.x); a.h[1]=f2bf(a0.y); a.h[2]=f2bf(a0.z); a.h[3]=f2bf(a0.w);
        a.h[4]=f2bf(a1.x); a.h[5]=f2bf(a1.y); a.h[6]=f2bf(a1.z); a.h[7]=f2bf(a1.w);
        a.h[8]=f2bf(a2.x); a.h[9]=f2bf(a2.y); a.h[10]=f2bf(a2.z); a.h[11]=f2bf(a2.w);
        a.h[12]=f2bf(a3.x); a.h[13]=f2bf(a3.y); a.h[14]=f2bf(a3.z); a.h[15]=f2bf(a3.w);
      }
      BFPack16 b;
      {
        const float* brow = B + (long long)(n0 + bn) * ldb + k0 + bkb;
        const float4 b0 = *(const float4*)(brow);
        const float4 b1 = *(const float4*)(brow + 4);
        const float4 b2 = *(const float4*)(brow + 16);
        const float4 b3 = *(const float4*)(brow + 20);
        b.h[0]=f2bf(b0.x); b.h[1]=f2bf(b0.y); b.h[2]=f2bf(b0.z); b.h[3]=f2bf(b0.w);
        b.h[4]=f2bf(b1.x); b.h[5]=f2bf(b1.y); b.h[6]=f2bf(b1.z); b.h[7]=f2bf(b1.w);
        b.h[8]=f2bf(b2.x); b.h[9]=f2bf(b2.y); b.h[10]=f2bf(b2.z); b.h[11]=f2bf(b2.w);
        b.h[12]=f2bf(b3.x); b.h[13]=f2bf(b3.y); b.h[14]=f2bf(b3.z); b.h[15]=f2bf(b3.w);
        __builtin_prefetch(brow + 32, 0, 1);   // global_prefetch next K slab
      }
      acc = __builtin_amdgcn_wmma_f32_16x16x32_bf16(
          false, a.v, false, b.v, (short)0, acc, false, false);
    }
  } else {
    // ---- NN: async-stage 32x64 B tile into LDS, double-buffered ----------
    // 68-float row pitch (272B = 17*16B) keeps every b128 dest 16B-aligned.
    __shared__ float Bs[2][32][68];
    const int srow = threadIdx.x >> 3;          // 0..31 (k within tile)
    const int scol = (threadIdx.x & 7) * 8;     // 0,8,..,56 (n within tile)
    const int colL = ((wave & 3) << 4) + bn;    // this lane's n within tile

    // stage tile [k0..k0+32) x [nblk..nblk+64) into buffer `buf`
    auto stage = [&](int k0, int buf) {
      const float* src = B + (long long)(k0 + srow) * ldb + nblk + scol;
      unsigned off = (unsigned)(unsigned long long)&Bs[buf][srow][scol];
      async_copy_b128(off,      src);
      async_copy_b128(off + 16, src + 4);
    };

    stage(0, 0);
    for (int k0 = 0; k0 < K; k0 += 32) {
      const int buf = (k0 >> 5) & 1;
      wait_async0();        // tile k0 landed (issued last iteration)
      __syncthreads();      // visible to all waves; prev reads of buf^1 done
      if (k0 + 32 < K) stage(k0 + 32, buf ^ 1);  // overlap copy with WMMA

      BFPack16 a;
      {
        const float4 a0 = *(const float4*)(arow + k0 + akb);
        const float4 a1 = *(const float4*)(arow + k0 + akb + 4);
        const float4 a2 = *(const float4*)(arow + k0 + akb + 16);
        const float4 a3 = *(const float4*)(arow + k0 + akb + 20);
        a.h[0]=f2bf(a0.x); a.h[1]=f2bf(a0.y); a.h[2]=f2bf(a0.z); a.h[3]=f2bf(a0.w);
        a.h[4]=f2bf(a1.x); a.h[5]=f2bf(a1.y); a.h[6]=f2bf(a1.z); a.h[7]=f2bf(a1.w);
        a.h[8]=f2bf(a2.x); a.h[9]=f2bf(a2.y); a.h[10]=f2bf(a2.z); a.h[11]=f2bf(a2.w);
        a.h[12]=f2bf(a3.x); a.h[13]=f2bf(a3.y); a.h[14]=f2bf(a3.z); a.h[15]=f2bf(a3.w);
      }
      BFPack16 b;
#pragma unroll
      for (int e = 0; e < 16; ++e)
        b.h[e] = f2bf(Bs[buf][bkb + e + (e >= 8 ? 8 : 0)][colL]);

      acc = __builtin_amdgcn_wmma_f32_16x16x32_bf16(
          false, a.v, false, b.v, (short)0, acc, false, false);
    }
  }

  const float bv = bias ? bias[n0 + cn] : 0.0f;
#pragma unroll
  for (int v = 0; v < 8; ++v)
    C[(long long)(m0 + cm + v) * ldc + (n0 + cn)] = acc[v] * alpha + bv;
}

// In-place row softmax; one 256-thread (8-wave) block per row.
__global__ __launch_bounds__(256) void softmax_rows(float* __restrict__ S, int n) {
  float* row = S + (long long)blockIdx.x * n;
  const int tid = threadIdx.x;
  __shared__ float sred[8];
  __shared__ float sb;

  float m = -3.402823466e38f;
  for (int i = tid; i < n; i += 256) m = fmaxf(m, row[i]);
#pragma unroll
  for (int off = 16; off > 0; off >>= 1) m = fmaxf(m, __shfl_xor(m, off, 32));
  if ((tid & 31) == 0) sred[tid >> 5] = m;
  __syncthreads();
  if (tid < 8) {
    float v = sred[tid];
#pragma unroll
    for (int off = 4; off > 0; off >>= 1) v = fmaxf(v, __shfl_xor(v, off, 32));
    if (tid == 0) sb = v;
  }
  __syncthreads();
  m = sb;

  float acc = 0.0f;
  for (int i = tid; i < n; i += 256) {
    float e = __expf(row[i] - m);
    row[i] = e;
    acc += e;
  }
#pragma unroll
  for (int off = 16; off > 0; off >>= 1) acc += __shfl_xor(acc, off, 32);
  if ((tid & 31) == 0) sred[tid >> 5] = acc;
  __syncthreads();
  if (tid < 8) {
    float v = sred[tid];
#pragma unroll
    for (int off = 4; off > 0; off >>= 1) v += __shfl_xor(v, off, 32);
    if (tid == 0) sb = v;
  }
  __syncthreads();
  const float inv = 1.0f / sb;
  for (int i = tid; i < n; i += 256) row[i] *= inv;
}

extern "C" void kernel_launch(void* const* d_in, const int* in_sizes, int n_in,
                              void* d_out, int out_size, void* d_ws, size_t ws_size,
                              hipStream_t stream) {
  (void)in_sizes; (void)n_in; (void)out_size; (void)ws_size;
  const float* hidden = (const float*)d_in[0];  // [32, 4096]
  const float* kv     = (const float*)d_in[1];  // [32, 4096, 512]
  const float* q_w    = (const float*)d_in[2];  // [4096, 4096]
  const float* q_b    = (const float*)d_in[3];  // [4096]
  const float* w_kc   = (const float*)d_in[4];  // [32, 128, 512]
  const float* w_vc   = (const float*)d_in[5];  // [32, 512, 128]
  const float* o_w    = (const float*)d_in[6];  // [4096, 4096]
  const float* o_b    = (const float*)d_in[7];  // [4096]
  float* out = (float*)d_out;                   // [32, 4096]

  float* ws   = (float*)d_ws;
  float* q    = ws;                 // [32, 4096]            131072 f
  float* qab  = q    + 131072;      // [32, 32, 512]         524288 f
  float* sc   = qab  + 524288;      // [32, 32, 4096]       4194304 f
  float* ctx  = sc   + 4194304;     // [32, 32, 512]         524288 f
  float* oh   = ctx  + 524288;      // [32, 4096]            131072 f
                                    // total ~21 MB of d_ws

  const dim3 blk(256, 1, 1);
  const float SCALE = 0.08838834764831845f;  // 1/sqrt(128)

  // 1) q = hidden @ q_w^T + q_b                       [NT] M=32 N=4096 K=4096
  gemm_wmma<true ><<<dim3(64, 1, 1),  blk, 0, stream>>>(
      hidden, 4096, 0, q_w, 4096, 0, q, 4096, 0, q_b, 32, 4096, 4096, 1.0f);

  // 2) q_abs[b,h,:] = q[b,h*128:] @ w_kc[h]           [NN] z=h, M=32 N=512 K=128
  gemm_wmma<false><<<dim3(8, 1, 32),  blk, 0, stream>>>(
      q, 4096, 128, w_kc, 512, 65536, qab, 16384, 512, nullptr, 32, 512, 128, 1.0f);

  // 3) scores[b] = q_abs[b] @ kv[b]^T * SCALE         [NT] z=b, M=32 N=4096 K=512
  gemm_wmma<true ><<<dim3(64, 1, 32), blk, 0, stream>>>(
      qab, 512, 16384, kv, 512, 2097152, sc, 4096, 131072, nullptr, 32, 4096, 512, SCALE);

  // 4) softmax over S (1024 rows of 4096)
  softmax_rows<<<dim3(1024, 1, 1), blk, 0, stream>>>(sc, 4096);

  // 5) ctx[b] = attn[b] @ kv[b]                       [NN] z=b, M=32 N=512 K=4096
  gemm_wmma<false><<<dim3(8, 1, 32),  blk, 0, stream>>>(
      sc, 4096, 131072, kv, 512, 2097152, ctx, 512, 16384, nullptr, 32, 512, 4096, 1.0f);

  // 6) oh[b,h*128:] = ctx[b,h,:] @ w_vc[h]            [NN] z=h, M=32 N=128 K=512
  gemm_wmma<false><<<dim3(2, 1, 32),  blk, 0, stream>>>(
      ctx, 16384, 512, w_vc, 128, 65536, oh, 4096, 128, nullptr, 32, 128, 512, 1.0f);

  // 7) out = oh @ o_w^T + o_b                         [NT] M=32 N=4096 K=4096
  gemm_wmma<true ><<<dim3(64, 1, 1),  blk, 0, stream>>>(
      oh, 4096, 0, o_w, 4096, 0, out, 4096, 0, o_b, 32, 4096, 4096, 1.0f);
}
